// MaxCutLiftLayer_14448269983745
// MI455X (gfx1250) — matrix operand order repaired and checked by reading, not compiled, not run on hardware
//
#include <hip/hip_runtime.h>

#define N_NODES 100000
#define N_EDGES 600000
#define D       128
#define TWO_D   256
#define N_TILES 6250          // N_NODES / 16 (exact)
#define WAVES_PER_BLK 4

typedef __attribute__((ext_vector_type(16))) _Float16 v16h;
typedef __attribute__((ext_vector_type(8)))  _Float16 v8h;
typedef __attribute__((ext_vector_type(4)))  _Float16 v4h;
typedef __attribute__((ext_vector_type(8)))  float    v8f;

// ---------------------------------------------------------------------------
// k_init: zero the grads accumulator (lives in d_out, exactly N*D f32) and
// convert W (D x 2D, f32) to f16 into d_ws (64 KB, stays hot in WGP$/L2).
// ---------------------------------------------------------------------------
__global__ void k_init(float* grads, const float* __restrict__ W,
                       _Float16* __restrict__ Wh) {
  int i = blockIdx.x * blockDim.x + threadIdx.x;
  if (i < (N_NODES * D) / 4) {
    ((float4*)grads)[i] = make_float4(0.f, 0.f, 0.f, 0.f);
  }
  if (i < (D * TWO_D) / 4) {
    float4 w = ((const float4*)W)[i];
    v4h h;
    h[0] = (_Float16)w.x; h[1] = (_Float16)w.y;
    h[2] = (_Float16)w.z; h[3] = (_Float16)w.w;
    *(v4h*)(Wh + i * 4) = h;
  }
}

// ---------------------------------------------------------------------------
// k_scatter: one wave32 per edge. Each lane handles 4 consecutive floats of
// the 128-wide row: float4 gather of x[src], scale by edge weight, HW float
// atomics (global_atomic_add_f32) into grads[dst].
// ---------------------------------------------------------------------------
__global__ void k_scatter(const float* __restrict__ x,
                          const int* __restrict__ ei,
                          const float* __restrict__ ew,
                          float* grads) {
  long long g = (long long)blockIdx.x * blockDim.x + threadIdx.x;
  int e    = (int)(g >> 5);
  int lane = (int)(g & 31);
  if (e >= N_EDGES) return;
  int   src = ei[e];
  int   dst = ei[N_EDGES + e];
  float w   = ew[e];
  const float4 xv = *(const float4*)(x + (size_t)src * D + lane * 4);
  float* gp = grads + (size_t)dst * D + lane * 4;
  unsafeAtomicAdd(gp + 0, xv.x * w);
  unsafeAtomicAdd(gp + 1, xv.y * w);
  unsafeAtomicAdd(gp + 2, xv.z * w);
  unsafeAtomicAdd(gp + 3, xv.w * w);
}

// ---------------------------------------------------------------------------
// k_fused: one wave per 16-node tile.
//   Phase 1: normalize grads rows, stage A = [x | g/||g||] as f16 in LDS.
//   Phase 2: C(16x128) = A(16x256) @ W^T via 8x8 v_wmma_f32_16x16x32_f16.
//   Phase 3: + bias, per-row L2 normalize (half-wave shfl reduction), store.
// WMMA layouts (CDNA5 ISA):
//   A 16x32 f16: lane&15 = row M; half=lane>>4; elems[0..7]=K0+8h+j,
//                elems[8..15]=K0+16+8h+j  -> two ds_load_b128 per frag.
//   B 32x16 f16: lane&15 = col N; elems[0..7]=W[N][K0+8h+j], elems[8..15]=W[N][K0+16+8h+j]
//                -> two 16-byte loads from the f16 W copy (same k order as A).
//   C 16x16 f32: lane&15 = N; VGPR v -> M = v + 8*half.
// ---------------------------------------------------------------------------
__global__ void __launch_bounds__(32 * WAVES_PER_BLK)
k_fused(const float* __restrict__ x,
        const float* grads,                 // aliases `out` (d_out): no restrict
        const _Float16* __restrict__ Wh,
        const float* __restrict__ bvec,
        float* out) {
  __shared__ _Float16 ldsA[WAVES_PER_BLK * 16 * TWO_D];   // 32 KB

  const int lane = threadIdx.x & 31;
  const int wid  = threadIdx.x >> 5;
  const int tile = blockIdx.x * WAVES_PER_BLK + wid;
  const bool active = (tile < N_TILES);                   // wave-uniform
  _Float16* aBase = &ldsA[wid * 16 * TWO_D];

  // ---- Phase 1: build A tile in LDS (2 lanes per node row) ----
  if (active) {
    const int r    = lane >> 1;        // row within tile
    const int part = lane & 1;         // which 64-column half
    const int node = tile * 16 + r;
    const float4* gsrc = (const float4*)(grads + (size_t)node * D + part * 64);
    const float4* xsrc = (const float4*)(x     + (size_t)node * D + part * 64);

    float4 gv[16];
    float ss = 0.f;
#pragma unroll
    for (int j = 0; j < 16; ++j) {
      gv[j] = gsrc[j];
      ss += gv[j].x * gv[j].x + gv[j].y * gv[j].y +
            gv[j].z * gv[j].z + gv[j].w * gv[j].w;
    }
    ss += __shfl_xor(ss, 1, 32);                         // combine row halves
    const float inv = 1.f / fmaxf(sqrtf(ss), 1e-12f);

    _Float16* axp = aBase + r * TWO_D + part * 64;       // x part
    _Float16* agp = axp + D;                             // normalized grads part
#pragma unroll
    for (int j = 0; j < 16; ++j) {
      float4 xv = xsrc[j];
      v4h hx;
      hx[0] = (_Float16)xv.x; hx[1] = (_Float16)xv.y;
      hx[2] = (_Float16)xv.z; hx[3] = (_Float16)xv.w;
      *(v4h*)(axp + j * 4) = hx;
      v4h hg;
      hg[0] = (_Float16)(gv[j].x * inv); hg[1] = (_Float16)(gv[j].y * inv);
      hg[2] = (_Float16)(gv[j].z * inv); hg[3] = (_Float16)(gv[j].w * inv);
      *(v4h*)(agp + j * 4) = hg;
    }
  }
  __syncthreads();
  if (!active) return;

  // ---- Phase 2: WMMA GEMM ----
  const int ll = lane & 15;
  const int hf = lane >> 4;
  const _Float16* arow = aBase + ll * TWO_D;

  float bv[8];
#pragma unroll
  for (int nt = 0; nt < 8; ++nt) bv[nt] = bvec[nt * 16 + ll];

  v8f acc[8] = {};
#pragma unroll
  for (int kt = 0; kt < 8; ++kt) {
    const int K0 = kt * 32;
    v8h alo = *(const v8h*)(arow + K0 + 8 * hf);
    v8h ahi = *(const v8h*)(arow + K0 + 16 + 8 * hf);
    v16h a = __builtin_shufflevector(alo, ahi, 0, 1, 2, 3, 4, 5, 6, 7,
                                     8, 9, 10, 11, 12, 13, 14, 15);
#pragma unroll
    for (int nt = 0; nt < 8; ++nt) {
      const _Float16* bp = Wh + (size_t)(nt * 16 + ll) * TWO_D + K0 + 8 * hf;
      v8h b0 = *(const v8h*)bp;
      v8h b1 = *(const v8h*)(bp + 16);
      v16h bb = __builtin_shufflevector(b0, b1, 0, 1, 2, 3, 4, 5, 6, 7,
                                        8, 9, 10, 11, 12, 13, 14, 15);
      acc[nt] = __builtin_amdgcn_wmma_f32_16x16x32_f16(
          false, a, false, bb, (short)0, acc[nt], false, false);
    }
  }

  // ---- Phase 3: bias, L2 normalize rows, store ----
#pragma unroll
  for (int nt = 0; nt < 8; ++nt)
#pragma unroll
    for (int v = 0; v < 8; ++v) acc[nt][v] += bv[nt];

  float p[8];
#pragma unroll
  for (int v = 0; v < 8; ++v) {
    float s = 0.f;
#pragma unroll
    for (int nt = 0; nt < 8; ++nt) s += acc[nt][v] * acc[nt][v];
    p[v] = s;
  }
#pragma unroll
  for (int m = 1; m < 16; m <<= 1)
#pragma unroll
    for (int v = 0; v < 8; ++v) p[v] += __shfl_xor(p[v], m, 32);

  float inv[8];
#pragma unroll
  for (int v = 0; v < 8; ++v) inv[v] = 1.f / fmaxf(sqrtf(p[v]), 1e-12f);

#pragma unroll
  for (int v = 0; v < 8; ++v) {
    float* orow = out + (size_t)(tile * 16 + v + 8 * hf) * D;
#pragma unroll
    for (int nt = 0; nt < 8; ++nt)
      orow[nt * 16 + ll] = acc[nt][v] * inv[v];
  }
}

// ---------------------------------------------------------------------------
extern "C" void kernel_launch(void* const* d_in, const int* in_sizes, int n_in,
                              void* d_out, int out_size, void* d_ws, size_t ws_size,
                              hipStream_t stream) {
  const float* x  = (const float*)d_in[0];
  const int*   ei = (const int*)d_in[1];
  const float* ew = (const float*)d_in[2];
  const float* W  = (const float*)d_in[3];
  const float* b  = (const float*)d_in[4];

  float*     out   = (float*)d_out;
  float*     grads = (float*)d_out;              // reused as accumulator
  _Float16*  Wh    = (_Float16*)d_ws;            // 64 KB f16 copy of W

  // 1) zero grads + convert W
  {
    int threads = (N_NODES * D) / 4;             // 3.2M
    k_init<<<(threads + 255) / 256, 256, 0, stream>>>(grads, W, Wh);
  }
  // 2) scatter-add messages (one wave per edge)
  {
    long long threads = (long long)N_EDGES * 32;
    k_scatter<<<(int)((threads + 255) / 256), 256, 0, stream>>>(x, ei, ew, grads);
  }
  // 3) normalize + concat + GEMM(WMMA) + bias + normalize
  {
    int blocks = (N_TILES + WAVES_PER_BLK - 1) / WAVES_PER_BLK;
    k_fused<<<blocks, 32 * WAVES_PER_BLK, 0, stream>>>(x, grads, Wh, b, out);
  }
}
